// PolicyNetRSNNPB_cstork_24034636989095
// MI455X (gfx1250) — matrix operand order repaired and checked by reading, not compiled, not run on hardware
//
#include <hip/hip_runtime.h>

typedef __attribute__((ext_vector_type(16))) __bf16 v16bf;
typedef __attribute__((ext_vector_type(8)))  __bf16 v8bf;
typedef __attribute__((ext_vector_type(4)))  __bf16 v4bf;
typedef __attribute__((ext_vector_type(8)))  float  v8f;

#define NB     512
#define HID    2048
#define INDIM  128
#define OUTD   64
#define MICRO  128

#define DCY_M 0.8187307530779818f
#define DCY_S 0.6065306597126334f
#define SCL_M 0.1812692469220182f

// ---------------- weight/input conversion (fp32 -> bf16, transposed) ----------------
__global__ void convert_kernel(const float* __restrict__ state,
                               const float* __restrict__ target,
                               const float* __restrict__ Wi,
                               const float* __restrict__ Vr,
                               const float* __restrict__ Wf,
                               const float* __restrict__ Wo,
                               __bf16* __restrict__ WiT,
                               __bf16* __restrict__ VrT,
                               __bf16* __restrict__ WfT,
                               __bf16* __restrict__ WoT,
                               __bf16* __restrict__ xbf) {
  unsigned i = blockIdx.x * 256u + threadIdx.x;
  const unsigned nWiT = 2048u * 128u;       // 262144
  const unsigned nBig = 2048u * 2048u;      // 4194304
  const unsigned nWoT = 64u * 2048u;        // 131072
  const unsigned nX   = 64u * 512u * 128u;  // 4194304
  if (i < nWiT) {                                  // WiT[n][k] = Wi[k][n], k<128
    unsigned n = i / 128u, k = i % 128u;
    WiT[i] = (__bf16)Wi[k * 2048u + n];
    return;
  }
  i -= nWiT;
  if (i < nBig) {                                  // VrT[n][k] = Vr[k][n]
    unsigned n = i / 2048u, k = i % 2048u;
    VrT[i] = (__bf16)Vr[k * 2048u + n];
    return;
  }
  i -= nBig;
  if (i < nBig) {                                  // WfT[n][k] = Wf[k][n]
    unsigned n = i / 2048u, k = i % 2048u;
    WfT[i] = (__bf16)Wf[k * 2048u + n];
    return;
  }
  i -= nBig;
  if (i < nWoT) {                                  // WoT[n][k] = Wo[k][n], n<64
    unsigned n = i / 2048u, k = i % 2048u;
    WoT[i] = (__bf16)Wo[k * 64u + n];
    return;
  }
  i -= nWoT;
  if (i < nX) {                                    // xbf[t][n][0:64]=state, [64:128]=target
    unsigned t = i / (512u * 128u);
    unsigned r = i % (512u * 128u);
    unsigned n = r / 128u, c = r % 128u;
    float v = (c < 64u) ? state[(t * 512u + n) * 64u + c]
                        : target[(t * 512u + n) * 64u + (c - 64u)];
    xbf[i] = (__bf16)v;
  }
}

// ---------------- zero state region ----------------
__global__ void zero_kernel(float4* __restrict__ p, unsigned n16) {
  unsigned i = blockIdx.x * 256u + threadIdx.x;
  if (i < n16) p[i] = make_float4(0.f, 0.f, 0.f, 0.f);
}

// ---------------- elementwise LIF state / spike kernel (x4 vectorized) ----------------
__global__ void state_kernel(const float* __restrict__ syn1, float* __restrict__ mem1,
                             const __bf16* __restrict__ s1rd, __bf16* __restrict__ s1wr,
                             const float* __restrict__ syn2, float* __restrict__ mem2,
                             const __bf16* __restrict__ s2rd, __bf16* __restrict__ s2wr,
                             const float* __restrict__ synr, float* __restrict__ memr,
                             float* __restrict__ out_mu, float* __restrict__ out_lv,
                             int t) {
  unsigned i = blockIdx.x * 256u + threadIdx.x;      // each thread: 4 elements
  const unsigned L4 = (NB * HID) / 4u;               // 262144
  if (i < L4) {
    float4 s = ((const float4*)syn1)[i];             // carry nsyn1(t-1)
    float4 m = ((const float4*)mem1)[i];
    v4bf  so = ((const v4bf*)s1rd)[i];
    float4 nm;
    nm.x = (DCY_M * m.x + SCL_M * s.x) * (1.0f - (float)so[0]);
    nm.y = (DCY_M * m.y + SCL_M * s.y) * (1.0f - (float)so[1]);
    nm.z = (DCY_M * m.z + SCL_M * s.z) * (1.0f - (float)so[2]);
    nm.w = (DCY_M * m.w + SCL_M * s.w) * (1.0f - (float)so[3]);
    ((float4*)mem1)[i] = nm;
    v4bf ns;
    ns[0] = (nm.x > 1.0f) ? (__bf16)1.0f : (__bf16)0.0f;
    ns[1] = (nm.y > 1.0f) ? (__bf16)1.0f : (__bf16)0.0f;
    ns[2] = (nm.z > 1.0f) ? (__bf16)1.0f : (__bf16)0.0f;
    ns[3] = (nm.w > 1.0f) ? (__bf16)1.0f : (__bf16)0.0f;
    ((v4bf*)s1wr)[i] = ns;
  } else if (i < 2u * L4) {
    unsigned e = i - L4;
    float4 s = ((const float4*)syn2)[e];
    float4 m = ((const float4*)mem2)[e];
    v4bf  so = ((const v4bf*)s2rd)[e];
    float4 nm;
    nm.x = (DCY_M * m.x + SCL_M * s.x) * (1.0f - (float)so[0]);
    nm.y = (DCY_M * m.y + SCL_M * s.y) * (1.0f - (float)so[1]);
    nm.z = (DCY_M * m.z + SCL_M * s.z) * (1.0f - (float)so[2]);
    nm.w = (DCY_M * m.w + SCL_M * s.w) * (1.0f - (float)so[3]);
    ((float4*)mem2)[e] = nm;
    v4bf ns;
    ns[0] = (nm.x > 1.0f) ? (__bf16)1.0f : (__bf16)0.0f;
    ns[1] = (nm.y > 1.0f) ? (__bf16)1.0f : (__bf16)0.0f;
    ns[2] = (nm.z > 1.0f) ? (__bf16)1.0f : (__bf16)0.0f;
    ns[3] = (nm.w > 1.0f) ? (__bf16)1.0f : (__bf16)0.0f;
    ((v4bf*)s2wr)[e] = ns;
  } else {
    unsigned e4 = i - 2u * L4;
    if (e4 < (unsigned)(NB * OUTD) / 4u) {
      float4 s = ((const float4*)synr)[e4];
      float4 m = ((const float4*)memr)[e4];
      float4 nm;
      nm.x = DCY_M * m.x + SCL_M * s.x;
      nm.y = DCY_M * m.y + SCL_M * s.y;
      nm.z = DCY_M * m.z + SCL_M * s.z;
      nm.w = DCY_M * m.w + SCL_M * s.w;
      ((float4*)memr)[e4] = nm;
      float vals[4] = {nm.x, nm.y, nm.z, nm.w};
      unsigned macro = (unsigned)t >> 1;
#pragma unroll
      for (int q = 0; q < 4; ++q) {
        unsigned e = e4 * 4u + q;
        unsigned n = e >> 6, c = e & 63u;
        float v = 0.5f * vals[q];
        float* dst = (c < 32u)
            ? (out_mu + ((size_t)macro * NB + n) * 32u + c)
            : (out_lv + ((size_t)macro * NB + n) * 32u + (c - 32u));
        if ((t & 1) == 0) *dst = v; else *dst += v;   // write then add: replay-safe
      }
    }
  }
}

// ---------------- WMMA GEMM helpers (32M x 64N wave tile) ----------------
__device__ __forceinline__ v16bf load_afrag(const __bf16* __restrict__ arow, int k, int h) {
  // A frag (16-bit 16x32): lane holds K = h*8..h*8+7 and 16+h*8..16+h*8+7
  v8bf a0 = *(const v8bf*)(arow + k + h * 8);
  v8bf a1 = *(const v8bf*)(arow + k + 16 + h * 8);
  v16bf a;
#pragma unroll
  for (int e = 0; e < 8; ++e) { a[e] = a0[e]; a[e + 8] = a1[e]; }
  return a;
}

__device__ __forceinline__ void wmma_accum2(v8f acc[8],
                                            const __bf16* __restrict__ A, int lda,
                                            const __bf16* __restrict__ BT, int ldb,
                                            int m0, int n0, int K, int lane) {
  const int l = lane & 15, h = lane >> 4;
  const __bf16* arow0 = A + (size_t)(m0 + l) * lda;
  const __bf16* arow1 = A + (size_t)(m0 + 16 + l) * lda;
  for (int k = 0; k < K; k += 32) {
    v16bf aLo = load_afrag(arow0, k, h);
    v16bf aHi = load_afrag(arow1, k, h);
#pragma unroll
    for (int j = 0; j < 4; ++j) {
      // B frag: lane = column n, K = h*8..h*8+7 and 16+h*8..16+h*8+7 (pre-transposed BT)
      const __bf16* bcol = BT + (size_t)(n0 + j * 16 + l) * ldb + k + h * 8;
      v8bf b0 = *(const v8bf*)(bcol);
      v8bf b1 = *(const v8bf*)(bcol + 16);
      v16bf b;
#pragma unroll
      for (int e = 0; e < 8; ++e) { b[e] = b0[e]; b[e + 8] = b1[e]; }
      acc[j]     = __builtin_amdgcn_wmma_f32_16x16x32_bf16(
          false, aLo, false, b, (short)0, acc[j], false, false);
      acc[4 + j] = __builtin_amdgcn_wmma_f32_16x16x32_bf16(
          false, aHi, false, b, (short)0, acc[4 + j], false, false);
    }
  }
}

// epilogue fuses the synaptic update: syn = DCY_S*syn + (GEMM + bias)
__device__ __forceinline__ void epilogue2(v8f acc[8], float* __restrict__ syn, int ldc,
                                          const float* __restrict__ bias,
                                          int m0, int n0, int lane) {
  const int l = lane & 15, h = lane >> 4;
#pragma unroll
  for (int j = 0; j < 4; ++j) {
    int col = n0 + j * 16 + l;
    float bv = bias[col];
#pragma unroll
    for (int r = 0; r < 8; ++r) {
      int row = m0 + r + 8 * h;                 // C/D layout: M = r + 8h, N = lane%16
      size_t idx0 = (size_t)row * ldc + col;
      size_t idx1 = (size_t)(row + 16) * ldc + col;
      syn[idx0] = DCY_S * syn[idx0] + (acc[j][r] + bv);
      syn[idx1] = DCY_S * syn[idx1] + (acc[4 + j][r] + bv);
    }
  }
}

// all three GEMMs of one step, wave-job decoded (1040 waves total)
__global__ void __launch_bounds__(128) matmul_kernel(
    const __bf16* __restrict__ xt,     // (512,128) bf16, current macro-step input
    const __bf16* __restrict__ WiT, const __bf16* __restrict__ VrT,
    const __bf16* __restrict__ WfT, const __bf16* __restrict__ WoT,
    const __bf16* __restrict__ s1_old, const __bf16* __restrict__ s1_new,
    const __bf16* __restrict__ s2_new,
    const float* __restrict__ bi, const float* __restrict__ bfv,
    const float* __restrict__ bo,
    float* __restrict__ syn1, float* __restrict__ syn2, float* __restrict__ synr) {
  int wid  = blockIdx.x * 4 + (threadIdx.x >> 5);
  int lane = threadIdx.x & 31;
  v8f zero{};
  v8f acc[8] = {zero, zero, zero, zero, zero, zero, zero, zero};

  if (wid < 512) {                  // GEMM1: I1 = xt@Wi + s1_old@Vr  (512x2048)
    int m0 = (wid >> 5) * 32, n0 = (wid & 31) * 64;
    wmma_accum2(acc, xt, INDIM, WiT, INDIM, m0, n0, INDIM, lane);
    wmma_accum2(acc, s1_old, HID, VrT, HID, m0, n0, HID, lane);
    epilogue2(acc, syn1, HID, bi, m0, n0, lane);
  } else if (wid < 1024) {          // GEMM2: I2 = ns1@Wf  (512x2048)
    int id = wid - 512;
    int m0 = (id >> 5) * 32, n0 = (id & 31) * 64;
    wmma_accum2(acc, s1_new, HID, WfT, HID, m0, n0, HID, lane);
    epilogue2(acc, syn2, HID, bfv, m0, n0, lane);
  } else {                          // GEMM3: Ir = ns2@Wo  (512x64)
    int id = wid - 1024;            // 0..15
    int m0 = id * 32, n0 = 0;
    wmma_accum2(acc, s2_new, HID, WoT, HID, m0, n0, HID, lane);
    epilogue2(acc, synr, OUTD, bo, m0, n0, lane);
  }
}

// ---------------- host launcher ----------------
extern "C" void kernel_launch(void* const* d_in, const int* in_sizes, int n_in,
                              void* d_out, int out_size, void* d_ws, size_t ws_size,
                              hipStream_t stream) {
  (void)in_sizes; (void)n_in; (void)out_size; (void)ws_size;
  const float* state  = (const float*)d_in[0];
  const float* target = (const float*)d_in[1];
  const float* Wi     = (const float*)d_in[2];
  const float* bi     = (const float*)d_in[3];
  const float* Vr     = (const float*)d_in[4];
  const float* Wf     = (const float*)d_in[5];
  const float* bfv    = (const float*)d_in[6];
  const float* Wo     = (const float*)d_in[7];
  const float* bo     = (const float*)d_in[8];
  float* out_mu = (float*)d_out;
  float* out_lv = out_mu + (size_t)64 * 512 * 32;

  char* ws = (char*)d_ws;
  size_t off = 0;
  auto carve = [&](size_t bytes) -> void* {
    void* p = (void*)(ws + off);
    off += (bytes + 255) & ~(size_t)255;
    return p;
  };
  __bf16* WiT = (__bf16*)carve((size_t)HID * INDIM * 2);
  __bf16* VrT = (__bf16*)carve((size_t)HID * HID * 2);
  __bf16* WfT = (__bf16*)carve((size_t)HID * HID * 2);
  __bf16* WoT = (__bf16*)carve((size_t)OUTD * HID * 2);
  __bf16* xbf = (__bf16*)carve((size_t)64 * NB * INDIM * 2);
  size_t zstart = off;                               // everything below is zero-inited
  __bf16* s1a = (__bf16*)carve((size_t)NB * HID * 2);
  __bf16* s1b = (__bf16*)carve((size_t)NB * HID * 2);
  __bf16* s2a = (__bf16*)carve((size_t)NB * HID * 2);
  __bf16* s2b = (__bf16*)carve((size_t)NB * HID * 2);
  float*  syn1 = (float*)carve((size_t)NB * HID * 4);
  float*  mem1 = (float*)carve((size_t)NB * HID * 4);
  float*  syn2 = (float*)carve((size_t)NB * HID * 4);
  float*  mem2 = (float*)carve((size_t)NB * HID * 4);
  float*  synr = (float*)carve((size_t)NB * OUTD * 4);
  float*  memr = (float*)carve((size_t)NB * OUTD * 4);
  size_t zbytes = off - zstart;

  // 1) convert weights (transposed) + inputs to bf16
  const unsigned convTotal = 262144u + 4194304u + 4194304u + 131072u + 4194304u;
  convert_kernel<<<(convTotal + 255u) / 256u, 256, 0, stream>>>(
      state, target, Wi, Vr, Wf, Wo, WiT, VrT, WfT, WoT, xbf);

  // 2) zero all carry state (fresh every launch: deterministic)
  unsigned zc = (unsigned)(zbytes / 16);
  zero_kernel<<<(zc + 255u) / 256u, 256, 0, stream>>>((float4*)(ws + zstart), zc);

  // 3) sequential scan: 128 micro-steps
  const unsigned stateThreads = (2u * NB * HID + NB * OUTD) / 4u;   // 532480
  for (int t = 0; t < MICRO; ++t) {
    const __bf16* s1rd = (t & 1) ? s1b : s1a;
    __bf16*       s1wr = (t & 1) ? s1a : s1b;
    const __bf16* s2rd = (t & 1) ? s2b : s2a;
    __bf16*       s2wr = (t & 1) ? s2a : s2b;

    state_kernel<<<(stateThreads + 255u) / 256u, 256, 0, stream>>>(
        syn1, mem1, s1rd, s1wr, syn2, mem2, s2rd, s2wr, synr, memr,
        out_mu, out_lv, t);

    if (t < MICRO - 1) {   // final step's currents are never consumed
      const __bf16* xt = xbf + (size_t)(t >> 1) * NB * INDIM;
      matmul_kernel<<<260, 128, 0, stream>>>(
          xt, WiT, VrT, WfT, WoT,
          s1rd /*carry spikes*/, s1wr /*new L1 spikes*/, s2wr /*new L2 spikes*/,
          bi, bfv, bo, syn1, syn2, synr);
    }
  }
}
